// AGCRN_30073361006781
// MI455X (gfx1250) — matrix operand order repaired and checked by reading, not compiled, not run on hardware
//
#include <hip/hip_runtime.h>
#include <math.h>

// ---------------------------------------------------------------------------
// AGCRN on MI455X (gfx1250): bf16 WMMA (v_wmma_f32_16x16x32_bf16) everywhere,
// f32 accumulate, fused GRU epilogues. All generated buffers are padded
// (Npad=2048 nodes, ccp=128 channels) so every global<->LDS staging loop is
// guard-free and 128-bit vectorized.
// ---------------------------------------------------------------------------

typedef __attribute__((ext_vector_type(16))) __bf16           v16bf;
typedef __attribute__((ext_vector_type(16))) unsigned short   v16u;
typedef __attribute__((ext_vector_type(8)))  unsigned short   v8us;
typedef __attribute__((ext_vector_type(8)))  float            v8f;

#define NPAD 2048
#define CCP  128

__device__ __forceinline__ unsigned short f2bf_u16(float f) {
    union { float f; unsigned u; } v; v.f = f;
    unsigned r = v.u + 0x7FFFu + ((v.u >> 16) & 1u);   // round-to-nearest-even
    return (unsigned short)(r >> 16);
}

// ---------------------------------------------------------------------------
// utility kernels
// ---------------------------------------------------------------------------
__global__ void agcrn_zero_f32(float* p, long long n) {
    long long i = (long long)blockIdx.x * blockDim.x + threadIdx.x;
    if (i < n) p[i] = 0.0f;
}

// S = relu(E E^T) into padded (NPAD x NPAD), zeros outside (N x N)
__global__ void agcrn_relu_eet(const float* __restrict__ E, float* __restrict__ S, int N) {
    long long i = (long long)blockIdx.x * blockDim.x + threadIdx.x;
    if (i >= (long long)NPAD * NPAD) return;
    int r = (int)(i / NPAD), c = (int)(i % NPAD);
    float out = 0.0f;
    if (r < N && c < N) {
        float acc = 0.0f;
        #pragma unroll
        for (int d = 0; d < 16; ++d) acc += E[r * 16 + d] * E[c * 16 + d];
        out = fmaxf(acc, 0.0f);
    }
    S[i] = out;
}

// row softmax over first N columns; bf16 out; pad rows/cols zero. grid = NPAD rows
__global__ void agcrn_row_softmax(const float* __restrict__ S, unsigned short* __restrict__ Abf, int N) {
    const int row = blockIdx.x;
    unsigned short* arow = Abf + (long long)row * NPAD;
    if (row >= N) {
        for (int j = threadIdx.x; j < NPAD; j += 256) arow[j] = 0;
        return;
    }
    const float* s = S + (long long)row * NPAD;
    __shared__ float red[256];
    float m = -1e30f;
    for (int j = threadIdx.x; j < N; j += 256) m = fmaxf(m, s[j]);
    red[threadIdx.x] = m; __syncthreads();
    for (int st = 128; st > 0; st >>= 1) {
        if (threadIdx.x < st) red[threadIdx.x] = fmaxf(red[threadIdx.x], red[threadIdx.x + st]);
        __syncthreads();
    }
    m = red[0]; __syncthreads();
    float sum = 0.0f;
    for (int j = threadIdx.x; j < N; j += 256) sum += __expf(s[j] - m);
    red[threadIdx.x] = sum; __syncthreads();
    for (int st = 128; st > 0; st >>= 1) {
        if (threadIdx.x < st) red[threadIdx.x] += red[threadIdx.x + st];
        __syncthreads();
    }
    float inv = 1.0f / red[0];
    for (int j = threadIdx.x; j < NPAD; j += 256)
        arow[j] = (j < N) ? f2bf_u16(__expf(s[j] - m) * inv) : (unsigned short)0;
}

// A2 = bf16(2*P - I) over padded
__global__ void agcrn_make_a2(const float* __restrict__ P, unsigned short* __restrict__ A2, int N) {
    long long i = (long long)blockIdx.x * blockDim.x + threadIdx.x;
    if (i >= (long long)NPAD * NPAD) return;
    int r = (int)(i / NPAD), c = (int)(i % NPAD);
    A2[i] = f2bf_u16(2.0f * P[i] - ((r == c && r < N) ? 1.0f : 0.0f));
}

// per-node weights into padded layout (n, 3, CCP, O), zero for c >= cc
__global__ void agcrn_mix_pool_bf16(const float* __restrict__ E, const float* __restrict__ pool,
                                    unsigned short* __restrict__ out, int cc, int O) {
    const int n = blockIdx.y;
    const int Jp = 3 * CCP * O;
    int j = blockIdx.x * blockDim.x + threadIdx.x;
    if (j >= Jp) return;
    int k = j / (CCP * O);
    int rem = j % (CCP * O);
    int c = rem / O, o = rem % O;
    float acc = 0.0f;
    if (c < cc) {
        const int Jsrc = 3 * cc * O;
        long long pi = (long long)(k * cc + c) * O + o;
        #pragma unroll
        for (int d = 0; d < 16; ++d) acc += E[n * 16 + d] * pool[(long long)d * Jsrc + pi];
        out[(long long)n * Jp + j] = f2bf_u16(acc);
    } else {
        out[(long long)n * Jp + j] = 0;
    }
}

// per-node biases (f32)
__global__ void agcrn_mix_pool_f32(const float* __restrict__ E, const float* __restrict__ pool,
                                   float* __restrict__ out, int J) {
    int n = blockIdx.y;
    int j = blockIdx.x * blockDim.x + threadIdx.x;
    if (j >= J) return;
    float acc = 0.0f;
    #pragma unroll
    for (int d = 0; d < 16; ++d) acc += E[n * 16 + d] * pool[(long long)d * J + j];
    out[(long long)n * J + j] = acc;
}

// Xcat (B, NPAD, CCP) bf16:  [x_t | h | 0-pad]  or  [x_t | z*h | 0-pad], n < N only
__global__ void agcrn_concat(const float* __restrict__ xf, const unsigned short* __restrict__ xb,
                             long long x_boff, long long x_bstride, int cin,
                             const float* __restrict__ h, const float* __restrict__ z,
                             unsigned short* __restrict__ xcat, int N) {
    long long total = 64LL * N * CCP;
    long long i = (long long)blockIdx.x * blockDim.x + threadIdx.x;
    if (i >= total) return;
    int c = (int)(i % CCP);
    long long bn = i / CCP;
    int n = (int)(bn % N);
    int b = (int)(bn / N);
    unsigned short v = 0;
    if (c < cin) {
        long long xi = x_boff + (long long)b * x_bstride + (long long)n * cin + c;
        v = xf ? f2bf_u16(xf[xi]) : xb[xi];
    } else if (c < cin + 64) {
        long long hidx = ((long long)b * N + n) * 64 + (c - cin);
        float hv = h[hidx];
        v = f2bf_u16(z ? z[hidx] * hv : hv);
    }
    xcat[((long long)b * NPAD + n) * CCP + c] = v;
}

// y[b, t, n] = conv_w[t] . h[b,n,:] + conv_b[t]
__global__ void agcrn_projection(const float* __restrict__ h, const float* __restrict__ cw,
                                 const float* __restrict__ cb, float* __restrict__ out, int N) {
    long long i = (long long)blockIdx.x * blockDim.x + threadIdx.x;
    if (i >= 64LL * N) return;
    long long b = i / N, n = i % N;
    float hv[64];
    const float* hp = h + i * 64;
    #pragma unroll
    for (int j = 0; j < 64; ++j) hv[j] = hp[j];
    #pragma unroll
    for (int t = 0; t < 12; ++t) {
        float acc = cb[t];
        #pragma unroll
        for (int j = 0; j < 64; ++j) acc += cw[t * 64 + j] * hv[j];
        out[(b * 12 + t) * N + n] = acc;
    }
}

// ---------------------------------------------------------------------------
// fragment helpers (contiguous k order: lane half h holds k = 16h..16h+15)
// ---------------------------------------------------------------------------
__device__ __forceinline__ v16bf frag_from_rows(const unsigned short* base) {
    // A-matrix 16x32 frag: base -> &LDSrow[kb]; needs [0..15] contiguous
    v8us lo = *(const v8us*)(base);
    v8us hi = *(const v8us*)(base + 8);
    v16u au;
    #pragma unroll
    for (int j = 0; j < 8; ++j) { au[j] = lo[j]; au[8 + j] = hi[j]; }
    return __builtin_bit_cast(v16bf, au);
}
__device__ __forceinline__ v16bf frag_contig(const unsigned short* base) {
    // B-matrix 32x16 frag from transposed LDS: 16 contiguous K values
    v8us lo = *(const v8us*)(base);
    v8us hi = *(const v8us*)(base + 8);
    v16u bu;
    #pragma unroll
    for (int j = 0; j < 8; ++j) { bu[j] = lo[j]; bu[8 + j] = hi[j]; }
    return __builtin_bit_cast(v16bf, bu);
}

// ---------------------------------------------------------------------------
// Guard-free tiled bf16 WMMA GEMM: C[m,n] = sum_k A[m,k]*B[k,n]
//   M,K multiples of 64/32; Ncols multiple of 64. 128 threads, 64x64 tile.
//   batched over blockIdx.z. STORE_BF16 selects C dtype.
// ---------------------------------------------------------------------------
template <int STORE_BF16>
__global__ __launch_bounds__(128)
void agcrn_wmma_gemm(const unsigned short* __restrict__ A, long long lda, long long a_bstride,
                     const unsigned short* __restrict__ Bm, long long ldb, long long b_bstride,
                     void* __restrict__ Cm, long long ldc, long long c_bstride,
                     int K) {
    __shared__ unsigned short As[64][40];   // row-major 64x32 (+pad)
    __shared__ unsigned short Bt[64][40];   // TRANSPOSED: Bt[col][k] 64x32 (+pad)

    const int tid  = threadIdx.x;
    const int wave = tid >> 5;
    const int lane = tid & 31;
    const int lrow = lane & 15;
    const int lhi  = lane >> 4;
    const int bm = blockIdx.x * 64;
    const int bn = blockIdx.y * 64;
    const unsigned short* Ab = A  + (long long)blockIdx.z * a_bstride + (long long)bm * lda;
    const unsigned short* Bb = Bm + (long long)blockIdx.z * b_bstride + bn;
    const long long zC = (long long)blockIdx.z * c_bstride;
    const int wm = (wave >> 1) * 32;
    const int wn = (wave & 1) * 32;

    v8f acc[2][2] = {};

    for (int k0 = 0; k0 < K; k0 += 32) {
        // stage A tile 64x32: 256 x 16B chunks, vectorized, guard-free
        #pragma unroll
        for (int it = 0; it < 2; ++it) {
            int idx = tid + it * 128;
            int r = idx >> 2, c8 = (idx & 3) << 3;
            *(v8us*)&As[r][c8] = *(const v8us*)(Ab + (long long)r * lda + k0 + c8);
        }
        // stage B tile 32x64 transposed into Bt[col][k]
        #pragma unroll
        for (int it = 0; it < 2; ++it) {
            int idx = tid + it * 128;
            int r = idx >> 3, c8 = (idx & 7) << 3;
            v8us d = *(const v8us*)(Bb + (long long)(k0 + r) * ldb + c8);
            #pragma unroll
            for (int j = 0; j < 8; ++j) Bt[c8 + j][r] = d[j];
        }
        __syncthreads();

        const int kb  = lhi * 16;
        const int kb2 = lhi * 16;
        v16bf bfr[2];
        #pragma unroll
        for (int ni = 0; ni < 2; ++ni)
            bfr[ni] = frag_contig(&Bt[wn + ni * 16 + lrow][kb2]);
        #pragma unroll
        for (int mi = 0; mi < 2; ++mi) {
            v16bf af = frag_from_rows(&As[wm + mi * 16 + lrow][kb]);
            #pragma unroll
            for (int ni = 0; ni < 2; ++ni)
                acc[mi][ni] = __builtin_amdgcn_wmma_f32_16x16x32_bf16(
                    false, af, false, bfr[ni], (short)0, acc[mi][ni], false, false);
        }
        __syncthreads();
    }

    #pragma unroll
    for (int mi = 0; mi < 2; ++mi)
        #pragma unroll
        for (int ni = 0; ni < 2; ++ni) {
            int gc = bn + wn + ni * 16 + lrow;
            #pragma unroll
            for (int r = 0; r < 8; ++r) {
                int gr = bm + wm + mi * 16 + r + 8 * lhi;
                float v = acc[mi][ni][r];
                if (STORE_BF16)
                    ((unsigned short*)Cm)[zC + (long long)gr * ldc + gc] = f2bf_u16(v);
                else
                    ((float*)Cm)[zC + (long long)gr * ldc + gc] = v;
            }
        }
}

// ---------------------------------------------------------------------------
// Per-node weighted contraction with fused GRU epilogue.
//   OUT(64 x O) = sum_{k,c<kc_end} XG_k[b,n,c] * W[n,k,c,o] + bias[n,o]
//   MODE 0 (gate, O=128): sigmoid -> z (o<64), r (o>=64)
//   MODE 1 (update, O=64): tanh -> hc; h = r*h + (1-r)*hc; optional bf16 seq
// X buffers: (B, NPAD, CCP) bf16. W: (N, 3, CCP, O) bf16 (zero-padded rows).
// ---------------------------------------------------------------------------
template <int MODE>
__global__ __launch_bounds__(128)
void agcrn_node_apply(const unsigned short* __restrict__ xg0,
                      const unsigned short* __restrict__ xg1,
                      const unsigned short* __restrict__ xg2,
                      int kc_end,
                      const unsigned short* __restrict__ W,
                      const float* __restrict__ bias,
                      float* __restrict__ zbuf,
                      float* __restrict__ rbuf,
                      float* __restrict__ h,
                      unsigned short* __restrict__ hs_out,
                      long long hs_boff, long long hs_bstride,
                      int N) {
    constexpr int O  = (MODE == 0) ? 128 : 64;
    constexpr int OW = O / 4;
    constexpr int CT = OW / 16;

    __shared__ unsigned short Xs[64][40];      // Xs[b][c] 64x32 (+pad)
    __shared__ unsigned short Wt[O][40];       // TRANSPOSED: Wt[o][c] Ox32 (+pad)

    const int n = blockIdx.x;
    const int tid = threadIdx.x, wave = tid >> 5, lane = tid & 31;
    const int lrow = lane & 15, lhi = lane >> 4;

    v8f acc[4][CT] = {};

    const unsigned short* xgp[3] = { xg0, xg1, xg2 };
    for (int k = 0; k < 3; ++k) {
        const unsigned short* xg = xgp[k] + (long long)n * CCP;
        const unsigned short* Wk = W + ((long long)n * 3 + k) * CCP * O;
        for (int c0 = 0; c0 < kc_end; c0 += 32) {
            // stage X tile 64x32 (rows = batch b), guard-free vector loads
            #pragma unroll
            for (int it = 0; it < 2; ++it) {
                int idx = tid + it * 128;
                int b = idx >> 2, c8 = (idx & 3) << 3;
                *(v8us*)&Xs[b][c8] =
                    *(const v8us*)(xg + (long long)b * (NPAD * CCP) + c0 + c8);
            }
            // stage W tile 32xO transposed into Wt[o][c]
            #pragma unroll
            for (int it = 0; it < (O * 32) / (128 * 8); ++it) {
                int idx = tid + it * 128;
                int r = idx / (O / 8), o8 = (idx % (O / 8)) << 3;
                v8us d = *(const v8us*)(Wk + (long long)(c0 + r) * O + o8);
                #pragma unroll
                for (int j = 0; j < 8; ++j) Wt[o8 + j][r] = d[j];
            }
            __syncthreads();

            const int kb  = lhi * 16;
            const int kb2 = lhi * 16;
            v16bf bfv[CT];
            #pragma unroll
            for (int ct = 0; ct < CT; ++ct)
                bfv[ct] = frag_contig(&Wt[wave * OW + ct * 16 + lrow][kb2]);
            #pragma unroll
            for (int rt = 0; rt < 4; ++rt) {
                v16bf af = frag_from_rows(&Xs[rt * 16 + lrow][kb]);
                #pragma unroll
                for (int ct = 0; ct < CT; ++ct)
                    acc[rt][ct] = __builtin_amdgcn_wmma_f32_16x16x32_bf16(
                        false, af, false, bfv[ct], (short)0, acc[rt][ct], false, false);
            }
            __syncthreads();
        }
    }

    #pragma unroll
    for (int rt = 0; rt < 4; ++rt)
        #pragma unroll
        for (int ct = 0; ct < CT; ++ct) {
            const int o = wave * OW + ct * 16 + lrow;
            const float bs = bias[(long long)n * O + o];
            #pragma unroll
            for (int r8 = 0; r8 < 8; ++r8) {
                const int b = rt * 16 + r8 + 8 * lhi;
                float v = acc[rt][ct][r8] + bs;
                if constexpr (MODE == 0) {
                    float s = 1.0f / (1.0f + __expf(-v));
                    long long idx = ((long long)b * N + n) * 64;
                    if (o < 64) zbuf[idx + o] = s;
                    else        rbuf[idx + (o - 64)] = s;
                } else {
                    float hc = tanhf(v);
                    long long idx = ((long long)b * N + n) * 64 + o;
                    float rr = rbuf[idx];
                    float hn = rr * h[idx] + (1.0f - rr) * hc;
                    h[idx] = hn;
                    if (hs_out)
                        hs_out[hs_boff + (long long)b * hs_bstride + (long long)n * 64 + o] = f2bf_u16(hn);
                }
            }
        }
}

// ---------------------------------------------------------------------------
// host launcher
// ---------------------------------------------------------------------------
extern "C" void kernel_launch(void* const* d_in, const int* in_sizes, int n_in,
                              void* d_out, int out_size, void* d_ws, size_t ws_size,
                              hipStream_t stream) {
    (void)in_sizes; (void)n_in; (void)out_size; (void)ws_size;
    const float* source = (const float*)d_in[0];   // (64,12,2000,1)
    const float* E      = (const float*)d_in[1];   // (2000,16)
    const float* gwp0   = (const float*)d_in[2];   // (16,3,65,128)
    const float* gbp0   = (const float*)d_in[3];   // (16,128)
    const float* uwp0   = (const float*)d_in[4];   // (16,3,65,64)
    const float* ubp0   = (const float*)d_in[5];   // (16,64)
    const float* gwp1   = (const float*)d_in[6];   // (16,3,128,128)
    const float* gbp1   = (const float*)d_in[7];   // (16,128)
    const float* uwp1   = (const float*)d_in[8];   // (16,3,128,64)
    const float* ubp1   = (const float*)d_in[9];   // (16,64)
    const float* conv_w = (const float*)d_in[10];  // (12,64)
    const float* conv_b = (const float*)d_in[11];  // (12,)

    const int N = 2000, B = 64, T = 12;
    const int JpG = 3 * CCP * 128;   // padded per-node gate weight count
    const int JpU = 3 * CCP * 64;

    char* wsp = (char*)d_ws;
    auto alloc = [&](size_t bytes) -> char* {
        char* p = wsp;
        wsp += (bytes + 255) & ~(size_t)255;
        return p;
    };

    float*          Sbuf = (float*)alloc((size_t)NPAD * NPAD * 4);          // S, then P=A@A
    unsigned short* Abf  = (unsigned short*)alloc((size_t)NPAD * NPAD * 2);
    unsigned short* A2bf = (unsigned short*)alloc((size_t)NPAD * NPAD * 2);
    unsigned short* gW   = (unsigned short*)alloc((size_t)N * JpG * 2);
    unsigned short* uW   = (unsigned short*)alloc((size_t)N * JpU * 2);
    float*          gb   = (float*)alloc((size_t)N * 128 * 4);
    float*          ub   = (float*)alloc((size_t)N * 64 * 4);
    float*          hbuf = (float*)alloc((size_t)B * N * 64 * 4);
    float*          zbuf = (float*)alloc((size_t)B * N * 64 * 4);
    float*          rbuf = (float*)alloc((size_t)B * N * 64 * 4);
    unsigned short* Xcat = (unsigned short*)alloc((size_t)B * NPAD * CCP * 2);
    unsigned short* X1   = (unsigned short*)alloc((size_t)B * NPAD * CCP * 2);
    unsigned short* X2   = (unsigned short*)alloc((size_t)B * NPAD * CCP * 2);
    unsigned short* hs0  = (unsigned short*)alloc((size_t)B * T * N * 64 * 2);

    const long long xstrideB = (long long)NPAD * CCP;   // per-batch X stride

    // zero Xcat once (covers node-pad rows; concat rewrites n<N each step)
    {
        long long nf = (long long)B * NPAD * CCP * 2 / 4;
        agcrn_zero_f32<<<(unsigned)((nf + 255) / 256), 256, 0, stream>>>((float*)Xcat, nf);
    }

    // ---- adjacency: A = softmax(relu(E E^T)), A2 = 2*A@A - I (padded) ----
    {
        long long tot = (long long)NPAD * NPAD;
        agcrn_relu_eet<<<(unsigned)((tot + 255) / 256), 256, 0, stream>>>(E, Sbuf, N);
        agcrn_row_softmax<<<NPAD, 256, 0, stream>>>(Sbuf, Abf, N);
        dim3 g(NPAD / 64, NPAD / 64, 1);
        agcrn_wmma_gemm<0><<<g, 128, 0, stream>>>(Abf, NPAD, 0, Abf, NPAD, 0,
                                                  Sbuf, NPAD, 0, NPAD);
        agcrn_make_a2<<<(unsigned)((tot + 255) / 256), 256, 0, stream>>>(Sbuf, A2bf, N);
    }

    long long hsz = (long long)B * N * 64;
    dim3 gd(NPAD / 64, CCP / 64, B);

    for (int layer = 0; layer < 2; ++layer) {
        const int cin    = (layer == 0) ? 1 : 64;
        const int cc     = cin + 64;
        const int kc_end = (layer == 0) ? 96 : 128;   // cover cc, mult of 32
        const float* gwp = (layer == 0) ? gwp0 : gwp1;
        const float* gbp = (layer == 0) ? gbp0 : gbp1;
        const float* uwp = (layer == 0) ? uwp0 : uwp1;
        const float* ubp = (layer == 0) ? ubp0 : ubp1;

        agcrn_mix_pool_bf16<<<dim3((JpG + 255) / 256, N), 256, 0, stream>>>(E, gwp, gW, cc, 128);
        agcrn_mix_pool_bf16<<<dim3((JpU + 255) / 256, N), 256, 0, stream>>>(E, uwp, uW, cc, 64);
        agcrn_mix_pool_f32<<<dim3(1, N), 256, 0, stream>>>(E, gbp, gb, 128);
        agcrn_mix_pool_f32<<<dim3(1, N), 256, 0, stream>>>(E, ubp, ub, 64);
        agcrn_zero_f32<<<(unsigned)((hsz + 255) / 256), 256, 0, stream>>>(hbuf, hsz);

        const float*          xf = (layer == 0) ? source : nullptr;
        const unsigned short* xb = (layer == 0) ? nullptr : hs0;
        long long catTot = (long long)B * N * CCP;

        for (int t = 0; t < T; ++t) {
            long long xoff = (long long)t * N * cin;
            long long xbs  = (long long)T * N * cin;
            // ---- gate ----
            agcrn_concat<<<(unsigned)((catTot + 255) / 256), 256, 0, stream>>>(
                xf, xb, xoff, xbs, cin, hbuf, nullptr, Xcat, N);
            agcrn_wmma_gemm<1><<<gd, 128, 0, stream>>>(Abf,  NPAD, 0, Xcat, CCP, xstrideB,
                                                       X1, CCP, xstrideB, NPAD);
            agcrn_wmma_gemm<1><<<gd, 128, 0, stream>>>(A2bf, NPAD, 0, Xcat, CCP, xstrideB,
                                                       X2, CCP, xstrideB, NPAD);
            agcrn_node_apply<0><<<N, 128, 0, stream>>>(Xcat, X1, X2, kc_end, gW, gb,
                zbuf, rbuf, nullptr, nullptr, 0, 0, N);
            // ---- candidate + GRU update ----
            agcrn_concat<<<(unsigned)((catTot + 255) / 256), 256, 0, stream>>>(
                xf, xb, xoff, xbs, cin, hbuf, zbuf, Xcat, N);
            agcrn_wmma_gemm<1><<<gd, 128, 0, stream>>>(Abf,  NPAD, 0, Xcat, CCP, xstrideB,
                                                       X1, CCP, xstrideB, NPAD);
            agcrn_wmma_gemm<1><<<gd, 128, 0, stream>>>(A2bf, NPAD, 0, Xcat, CCP, xstrideB,
                                                       X2, CCP, xstrideB, NPAD);
            if (layer == 0)
                agcrn_node_apply<1><<<N, 128, 0, stream>>>(Xcat, X1, X2, kc_end, uW, ub,
                    nullptr, rbuf, hbuf, hs0, (long long)t * N * 64, (long long)T * N * 64, N);
            else
                agcrn_node_apply<1><<<N, 128, 0, stream>>>(Xcat, X1, X2, kc_end, uW, ub,
                    nullptr, rbuf, hbuf, nullptr, 0, 0, N);
        }
    }

    // ---- projection head: (B,12,N,1) ----
    {
        long long tot = (long long)B * N;
        agcrn_projection<<<(unsigned)((tot + 255) / 256), 256, 0, stream>>>(
            hbuf, conv_w, conv_b, (float*)d_out, N);
    }
}
